// CustomLSTMCell_49675591746256
// MI455X (gfx1250) — matrix operand order repaired and checked by reading, not compiled, not run on hardware
//
#include <hip/hip_runtime.h>
#include <stdint.h>

typedef __attribute__((ext_vector_type(16))) __bf16 v16bf;
typedef __attribute__((ext_vector_type(16))) float  v16f;
typedef __attribute__((ext_vector_type(8)))  float  v8f;
typedef unsigned short u16;
typedef unsigned int   u32;

#define DIM   2048          // B == I == H == 2048
#define BM    128           // block M tile
#define BN    32            // per-gate block N tile
#define KT    64            // K depth of each LDS buffer (2 WMMA sub-steps of 32)
#define NIT   ((2 * DIM) / KT)   // 64 iterations: first 32 over x/W_ih, rest over h/W_hh

union FragBF {
  uint4 q[2];
  v16bf v;
};

// Load 16 consecutive f32 at base[idx] (sgpr base + u32 vgpr offset -> GVS mode),
// convert to bf16 (v_cvt_pk_bf16_f32), store 32B to LDS.
__device__ __forceinline__ void stage16(const float* __restrict__ base, u32 idx, u16* dst) {
  v16f f = *(const v16f*)(base + idx);
  v16bf b = __builtin_convertvector(f, v16bf);
  *((v16bf*)dst) = b;
}

__device__ __forceinline__ float sigmoid_fast(float x) {
  return 1.0f / (1.0f + __expf(-x));
}
__device__ __forceinline__ float tanh_fast(float x) {
  return 1.0f - 2.0f / (__expf(2.0f * x) + 1.0f);
}

__global__ __launch_bounds__(256) void lstm_cell_wmma(
    const float* __restrict__ x,      const float* __restrict__ h_prev,
    const float* __restrict__ C_prev, const float* __restrict__ W_ih,
    const float* __restrict__ b_ih,   const float* __restrict__ W_hh,
    const float* __restrict__ b_hh,   float* __restrict__ out)
{
  // Double-buffered bf16 tiles: A = 128x64, B = 4 gates x 32 cols x 64 k (64KB total)
  __shared__ __align__(32) u16 sA[2][BM * KT];       // 16 KB each
  __shared__ __align__(32) u16 sB[2][4 * BN * KT];   // 16 KB each

  const int tid  = threadIdx.x;
  const int lane = tid & 31;
  const int wave = tid >> 5;
  const int wm   = wave & 3;   // 4 M sub-tiles of 32 rows
  const int wn   = wave >> 2;  // 2 N sub-tiles of 16 cols
  const int blockN = blockIdx.x * BN;
  const int blockM = blockIdx.y * BM;

  // staging coordinates (each thread moves 32 f32 of A and 32 f32 of W per iter)
  const int arow  = tid >> 1;          // 0..127
  const int hlf   = tid & 1;           // which 32-float half of a 64-wide row
  const int bgate = tid >> 6;          // 0..3
  const int brow  = (tid >> 1) & 31;   // 0..31 (local W row == output col)

  // u32 element offsets (max ~16.8M elements -> fits 32-bit, enables saddr+voffset)
  const u32 aOff = (u32)(blockM + arow) * DIM + (u32)hlf * 32u;
  const u32 wOff = (u32)(bgate * DIM + blockN + brow) * DIM + (u32)hlf * 32u;
  u16* const aDst = &sA[0][arow * KT + hlf * 32];          // buffer 0 base (buf adds BM*KT)
  u16* const wDst = &sB[0][(bgate * BN + brow) * KT + hlf * 32];

  v8f acc[2][4];
  {
    v8f z = {};
    #pragma unroll
    for (int i = 0; i < 2; ++i)
      #pragma unroll
      for (int g = 0; g < 4; ++g) acc[i][g] = z;
  }

  auto stageI = [&](int it, int buf) {
    const int ph  = it >> 5;                 // 0: x/W_ih, 1: h/W_hh
    const u32 kk  = (u32)(it & 31) * KT;     // k offset within the phase
    const float* As = ph ? h_prev : x;
    const float* Ws = ph ? W_hh   : W_ih;
    u16* ad = aDst + buf * (BM * KT);
    u16* wd = wDst + buf * (4 * BN * KT);
    stage16(As, aOff + kk,       ad);
    stage16(As, aOff + kk + 16u, ad + 16);
    stage16(Ws, wOff + kk,       wd);
    stage16(Ws, wOff + kk + 16u, wd + 16);
  };

  const int l15  = lane & 15;
  const int hiLn = lane >> 4;          // 0: low K half pattern, 1: high

  auto loadFrags = [&](int buf, int ks, FragBF (&fa)[2], FragBF (&fb)[4]) {
    // A fragments: 16x32 bf16; lanes 0-15 hold K0..7 & K16..23, lanes 16-31 K8..15 & K24..31
    #pragma unroll
    for (int mf = 0; mf < 2; ++mf) {
      const u16* base = &sA[buf][(wm * 32 + mf * 16 + l15) * KT + ks * 32 + hiLn * 8];
      fa[mf].q[0] = *(const uint4*)(base);
      fa[mf].q[1] = *(const uint4*)(base + 16);
    }
    // B fragments: 32x16 bf16; lanes 0-15 hold K0..7 & K16..23 of col n, lanes 16-31 K8..15 & K24..31
    #pragma unroll
    for (int g = 0; g < 4; ++g) {
      const u16* base = &sB[buf][(g * BN + wn * 16 + l15) * KT + ks * 32 + hiLn * 8];
      fb[g].q[0] = *(const uint4*)(base);
      fb[g].q[1] = *(const uint4*)(base + 16);
    }
  };

  auto wmmaAll = [&](FragBF (&fa)[2], FragBF (&fb)[4]) {
    #pragma unroll
    for (int g = 0; g < 4; ++g) {
      acc[0][g] = __builtin_amdgcn_wmma_f32_16x16x32_bf16(
          false, fa[0].v, false, fb[g].v, (short)0, acc[0][g], false, false);
      acc[1][g] = __builtin_amdgcn_wmma_f32_16x16x32_bf16(
          false, fa[1].v, false, fb[g].v, (short)0, acc[1][g], false, false);
    }
  };

  stageI(0, 0);
  for (int it = 0; it < NIT; ++it) {
    __syncthreads();
    const int buf = it & 1;

    // 1) sub-step 0 fragment loads first (lets WMMA wait on partial DScnt)
    FragBF fa0[2], fb0[4];
    loadFrags(buf, 0, fa0, fb0);

    // 2) prefetch iter+2 (two 128B lines per stream), stage iter+1
    if (it + 2 < NIT) {
      const int ph2 = (it + 2) >> 5;
      const u32 kk2 = (u32)((it + 2) & 31) * KT;
      const float* As2 = ph2 ? h_prev : x;
      const float* Ws2 = ph2 ? W_hh   : W_ih;
      __builtin_prefetch(As2 + aOff + kk2, 0, 1);
      __builtin_prefetch(As2 + aOff + kk2 + 32u, 0, 1);
      __builtin_prefetch(Ws2 + wOff + kk2, 0, 1);
      __builtin_prefetch(Ws2 + wOff + kk2 + 32u, 0, 1);
    }
    if (it + 1 < NIT) stageI(it + 1, buf ^ 1);

    // 3) matrix work: 2 sub-steps x (2 M-frags x 4 gates)
    wmmaAll(fa0, fb0);
    FragBF fa1[2], fb1[4];
    loadFrags(buf, 1, fa1, fb1);
    wmmaAll(fa1, fb1);
  }

  // ---- Fused epilogue: biases + gating + cell update, 6 outputs ----
  const int nCol = blockN + wn * 16 + l15;
  float bias[4];
  #pragma unroll
  for (int g = 0; g < 4; ++g)
    bias[g] = b_ih[g * DIM + nCol] + b_hh[g * DIM + nCol];

  const size_t P = (size_t)DIM * DIM;
  #pragma unroll
  for (int mf = 0; mf < 2; ++mf) {
    const int m0 = blockM + wm * 32 + mf * 16 + (hiLn ? 8 : 0);
    #pragma unroll
    for (int e = 0; e < 8; ++e) {
      const int m = m0 + e;
      const float gi = acc[mf][0][e] + bias[0];
      const float gf = acc[mf][1][e] + bias[1];
      const float gg = acc[mf][2][e] + bias[2];
      const float go = acc[mf][3][e] + bias[3];
      const float ig = sigmoid_fast(gi);
      const float fg = sigmoid_fast(gf);
      const float cg = tanh_fast(gg);
      const float og = sigmoid_fast(go);
      const float cp = C_prev[(size_t)m * DIM + nCol];
      const float nc = fg * cp + ig * cg;
      const float nh = og * tanh_fast(nc);
      const size_t o = (size_t)m * DIM + nCol;
      out[o]         = nh;   // new_h
      out[P + o]     = nc;   // new_C
      out[2 * P + o] = fg;   // forget_gate
      out[3 * P + o] = ig;   // input_gate
      out[4 * P + o] = cg;   // candidate_gate
      out[5 * P + o] = og;   // output_gate
    }
  }
}

extern "C" void kernel_launch(void* const* d_in, const int* in_sizes, int n_in,
                              void* d_out, int out_size, void* d_ws, size_t ws_size,
                              hipStream_t stream) {
  (void)in_sizes; (void)n_in; (void)out_size; (void)d_ws; (void)ws_size;
  const float* x      = (const float*)d_in[0];
  const float* h_prev = (const float*)d_in[1];
  const float* C_prev = (const float*)d_in[2];
  const float* W_ih   = (const float*)d_in[3];
  const float* b_ih   = (const float*)d_in[4];
  const float* W_hh   = (const float*)d_in[5];
  const float* b_hh   = (const float*)d_in[6];
  float* out = (float*)d_out;

  dim3 grid(DIM / BN, DIM / BM);  // 64 x 16 blocks
  lstm_cell_wmma<<<grid, 256, 0, stream>>>(x, h_prev, C_prev, W_ih, b_ih, W_hh, b_hh, out);
}
